// MixtralMoE_60215441490298
// MI455X (gfx1250) — matrix operand order repaired and checked by reading, not compiled, not run on hardware
//
#include <hip/hip_runtime.h>
#include <cstdint>

#define T_TOK 8192
#define HID   2048
#define INTERN 7168
#define NE    8

typedef __attribute__((ext_vector_type(16))) __bf16 v16bf;
typedef __attribute__((ext_vector_type(8)))  float  v8f;

union FragB { v16bf v; unsigned u[8]; };
union FragC { v8f  v; float    f[8]; };

// ---- fp32 -> bf16 (round-half-up; 1 add per src + 1 v_perm_b32 merge) ----
__device__ __forceinline__ unsigned pack_bf2(float a, float b) {
  const unsigned ua = __float_as_uint(a) + 0x8000u;
  const unsigned ub = __float_as_uint(b) + 0x8000u;
  // dst = { ub[31:16], ua[31:16] } : bytes {7,6} from ub(S0), bytes {3,2} from ua(S1)
  return __builtin_amdgcn_perm(ub, ua, 0x07060302u);
}
__device__ __forceinline__ unsigned short f2bf(float a) {
  return (unsigned short)((__float_as_uint(a) + 0x8000u) >> 16);
}

// ---- per-thread staging registers (16 elements = one LDS half-row) ----
struct TF4 { float4 f[4]; };   // 16 fp32
struct TU8 { uint4  u[2]; };   // 8 dwords = 16 bf16

__device__ __forceinline__ void fetch_f16x(TF4& r, const float* src) {
  const float4* s = (const float4*)src;
  #pragma unroll
  for (int j = 0; j < 4; ++j) r.f[j] = s[j];
}
__device__ __forceinline__ void zero_t(TF4& r) {
  #pragma unroll
  for (int j = 0; j < 4; ++j) r.f[j] = make_float4(0.f, 0.f, 0.f, 0.f);
}
__device__ __forceinline__ void stage_bf16(const TF4& r, unsigned* dst) {
  #pragma unroll
  for (int j = 0; j < 4; ++j) {
    dst[2 * j]     = pack_bf2(r.f[j].x, r.f[j].y);
    dst[2 * j + 1] = pack_bf2(r.f[j].z, r.f[j].w);
  }
}
__device__ __forceinline__ void fetch_u8(TU8& r, const unsigned* src) {
  const uint4* s = (const uint4*)src;
  r.u[0] = s[0]; r.u[1] = s[1];
}
__device__ __forceinline__ void stage_u8(const TU8& r, unsigned* dst) {
  const unsigned* p = (const unsigned*)&r;
  #pragma unroll
  for (int j = 0; j < 8; ++j) dst[j] = p[j];
}

// ---- WMMA fragment loads from LDS (pair-packed bf16, row stride 17 dwords) ----
// A 16x32 bf16: lane m = lane&15, half = lane>>4; VGPR j holds K-pair (j&3)+(j>>2)*8+half*4
__device__ __forceinline__ v16bf load_a_frag(const unsigned* As, int mBase, int lane) {
  const int m = mBase + (lane & 15);
  const int half = lane >> 4;
  FragB r;
  #pragma unroll
  for (int j = 0; j < 8; ++j) {
    const int p = (j & 3) + ((j >> 2) << 3) + half * 4;
    r.u[j] = As[m * 17 + p];
  }
  return r.v;
}
// B 32x16 bf16: lane n = lane&15, half = lane>>4; VGPR j holds K-pair (j&3)+(j>>2)*8+half*4
__device__ __forceinline__ v16bf load_b_frag(const unsigned* Bs, int nBase, int lane) {
  const int n = nBase + (lane & 15);
  const int half = lane >> 4;
  FragB r;
  #pragma unroll
  for (int j = 0; j < 8; ++j) r.u[j] = Bs[n * 17 + (j & 3) + ((j >> 2) << 3) + half * 4];
  return r.v;
}

// ---- one K-step of WMMAs (GEMM1: dual B for w1/w3) ----
__device__ __forceinline__ void mma_step_g1(
    const unsigned* As, const unsigned* Bs1, const unsigned* Bs3,
    int wM, int wN, int lane, v8f acc1[2][4], v8f acc3[2][4]) {
  v16bf af[2];
  #pragma unroll
  for (int mi = 0; mi < 2; ++mi) af[mi] = load_a_frag(As, wM + mi * 16, lane);
  #pragma unroll
  for (int ni = 0; ni < 4; ++ni) {
    const v16bf b1 = load_b_frag(Bs1, wN + ni * 16, lane);
    const v16bf b3 = load_b_frag(Bs3, wN + ni * 16, lane);
    #pragma unroll
    for (int mi = 0; mi < 2; ++mi) {
      acc1[mi][ni] = __builtin_amdgcn_wmma_f32_16x16x32_bf16(
          false, af[mi], false, b1, (short)0, acc1[mi][ni], false, false);
      acc3[mi][ni] = __builtin_amdgcn_wmma_f32_16x16x32_bf16(
          false, af[mi], false, b3, (short)0, acc3[mi][ni], false, false);
    }
  }
}
__device__ __forceinline__ void mma_step_g2(
    const unsigned* As, const unsigned* Bs,
    int wM, int wN, int lane, v8f acc[2][4]) {
  v16bf af[2];
  #pragma unroll
  for (int mi = 0; mi < 2; ++mi) af[mi] = load_a_frag(As, wM + mi * 16, lane);
  #pragma unroll
  for (int ni = 0; ni < 4; ++ni) {
    const v16bf b = load_b_frag(Bs, wN + ni * 16, lane);
    #pragma unroll
    for (int mi = 0; mi < 2; ++mi)
      acc[mi][ni] = __builtin_amdgcn_wmma_f32_16x16x32_bf16(
          false, af[mi], false, b, (short)0, acc[mi][ni], false, false);
  }
}

// ============ kernel 0: zero output + counts ============
__global__ void moe_zero(float* __restrict__ out, long n, int* __restrict__ counts) {
  long i = (long)blockIdx.x * blockDim.x + threadIdx.x;
  const long stride = (long)gridDim.x * blockDim.x;
  for (; i < n; i += stride) out[i] = 0.f;
  if (blockIdx.x == 0 && threadIdx.x < NE) counts[threadIdx.x] = 0;
}

// ============ kernel 1: router (one wave32 per token) ============
__global__ __launch_bounds__(256) void moe_router(
    const float* __restrict__ x, const float* __restrict__ gw,
    int* __restrict__ topi, float* __restrict__ topw)
{
  const int lane = threadIdx.x & 31;
  const int wid  = threadIdx.x >> 5;
  const int t = blockIdx.x * 8 + wid;
  const float* xr = x + (size_t)t * HID;

  float acc[NE];
  #pragma unroll
  for (int e = 0; e < NE; ++e) acc[e] = 0.f;
  for (int h = lane; h < HID; h += 32) {
    const float xv = xr[h];
    #pragma unroll
    for (int e = 0; e < NE; ++e) acc[e] += xv * gw[e * HID + h];
  }
  #pragma unroll
  for (int e = 0; e < NE; ++e)
    #pragma unroll
    for (int off = 16; off; off >>= 1) acc[e] += __shfl_xor(acc[e], off, 32);

  if (lane == 0) {
    float mx = acc[0];
    #pragma unroll
    for (int e = 1; e < NE; ++e) mx = fmaxf(mx, acc[e]);
    float p[NE];
    #pragma unroll
    for (int e = 0; e < NE; ++e) p[e] = __expf(acc[e] - mx);
    int i0 = 0; float p0 = p[0];
    #pragma unroll
    for (int e = 1; e < NE; ++e) if (p[e] > p0) { p0 = p[e]; i0 = e; }
    int i1 = -1; float p1 = -1.f;
    #pragma unroll
    for (int e = 0; e < NE; ++e) if (e != i0 && p[e] > p1) { p1 = p[e]; i1 = e; }
    const float inv = 1.f / (p0 + p1);
    topi[t * 2 + 0] = i0; topw[t * 2 + 0] = p0 * inv;
    topi[t * 2 + 1] = i1; topw[t * 2 + 1] = p1 * inv;
  }
}

// ============ kernel 2: build compacted per-expert token lists ============
__global__ void moe_build(const int* __restrict__ topi, const float* __restrict__ topw,
                          int* __restrict__ counts, int* __restrict__ rowtok,
                          float* __restrict__ roww)
{
  const int t = blockIdx.x * blockDim.x + threadIdx.x;
  if (t >= T_TOK) return;
  #pragma unroll
  for (int k = 0; k < 2; ++k) {
    const int e = topi[t * 2 + k];
    const int slot = atomicAdd(&counts[e], 1);
    rowtok[e * T_TOK + slot] = t;
    roww[e * T_TOK + slot]   = topw[t * 2 + k];
  }
}

// ============ kernel 3: GEMM1 (double-buffered): G = silu(Xg@w1^T)*(Xg@w3^T) ============
__global__ __launch_bounds__(256) void moe_gemm1(
    const float* __restrict__ x, const float* __restrict__ w1e,
    const float* __restrict__ w3e, const int* __restrict__ rowtok,
    const int* __restrict__ countp, unsigned short* __restrict__ G)
{
  const int count = *countp;
  const int m0 = blockIdx.y * 128;
  if (m0 >= count) return;
  const int n0 = blockIdx.x * 128;

  constexpr int BUF = 128 * 17;
  __shared__ int toks[128];
  __shared__ unsigned As [2 * BUF];
  __shared__ unsigned Bs1[2 * BUF];
  __shared__ unsigned Bs3[2 * BUF];

  const int tid = threadIdx.x;
  if (tid < 128) {
    const int m = m0 + tid;
    toks[tid] = (m < count) ? rowtok[m] : -1;
  }
  __syncthreads();

  const int lane = tid & 31;
  const int wid  = tid >> 5;
  const int wM = (wid >> 1) * 32;
  const int wN = (wid &  1) * 64;

  const v8f zero8 = {0.f, 0.f, 0.f, 0.f, 0.f, 0.f, 0.f, 0.f};
  v8f acc1[2][4], acc3[2][4];
  #pragma unroll
  for (int mi = 0; mi < 2; ++mi)
    #pragma unroll
    for (int ni = 0; ni < 4; ++ni) { acc1[mi][ni] = zero8; acc3[mi][ni] = zero8; }

  // loop-invariant per-thread addressing (t hoisted out of the K-loop)
  const int row  = tid >> 1;
  const int half = tid & 1;
  const int t    = toks[row];
  const float* xrow  = (t >= 0) ? (x + (size_t)t * HID + half * 16) : x;
  const float* w1row = w1e + (size_t)(n0 + row) * HID + half * 16;
  const float* w3row = w3e + (size_t)(n0 + row) * HID + half * 16;
  unsigned* const dA  = &As [row * 17 + half * 8];
  unsigned* const dB1 = &Bs1[row * 17 + half * 8];
  unsigned* const dB3 = &Bs3[row * 17 + half * 8];

  // prologue: fetch k=0, stage into buffer 0
  TF4 rA, rB1, rB3;
  if (t >= 0) fetch_f16x(rA, xrow); else zero_t(rA);
  fetch_f16x(rB1, w1row);
  fetch_f16x(rB3, w3row);
  stage_bf16(rA, dA); stage_bf16(rB1, dB1); stage_bf16(rB3, dB3);

  int buf = 0;
  for (int k0 = 32; k0 < HID; k0 += 32) {
    // issue next K-step's global loads before the barrier (latency overlap)
    if (t >= 0) fetch_f16x(rA, xrow + k0); else zero_t(rA);
    fetch_f16x(rB1, w1row + k0);
    fetch_f16x(rB3, w3row + k0);
    __builtin_prefetch((const void*)(w1row + k0 + 32), 0, 3);
    __builtin_prefetch((const void*)(w3row + k0 + 32), 0, 3);
    __syncthreads();                               // buf is fully staged
    mma_step_g1(As + buf * BUF, Bs1 + buf * BUF, Bs3 + buf * BUF,
                wM, wN, lane, acc1, acc3);
    const int nb = buf ^ 1;
    stage_bf16(rA, dA + nb * BUF);
    stage_bf16(rB1, dB1 + nb * BUF);
    stage_bf16(rB3, dB3 + nb * BUF);
    buf = nb;
  }
  __syncthreads();
  mma_step_g1(As + buf * BUF, Bs1 + buf * BUF, Bs3 + buf * BUF,
              wM, wN, lane, acc1, acc3);

  // epilogue: g = silu(a1) * a3 -> G (bf16). C layout: M = v + 8*(lane>>4), N = lane&15.
  const int lhalf = lane >> 4;
  const int lncol = lane & 15;
  #pragma unroll
  for (int mi = 0; mi < 2; ++mi) {
    #pragma unroll
    for (int ni = 0; ni < 4; ++ni) {
      FragC c1, c3;
      c1.v = acc1[mi][ni]; c3.v = acc3[mi][ni];
      #pragma unroll
      for (int v = 0; v < 8; ++v) {
        const float a = c1.f[v];
        const float g = (a / (1.f + __expf(-a))) * c3.f[v];
        const int mrow = m0 + wM + mi * 16 + lhalf * 8 + v;
        const int ncol = n0 + wN + ni * 16 + lncol;
        G[(size_t)mrow * INTERN + ncol] = f2bf(g);
      }
    }
  }
}

// ============ kernel 4: GEMM2 (double-buffered): out[tok] += w * (G @ w2^T) ============
__global__ __launch_bounds__(256) void moe_gemm2(
    const unsigned short* __restrict__ G, const float* __restrict__ w2e,
    const int* __restrict__ rowtok, const float* __restrict__ roww,
    const int* __restrict__ countp, float* __restrict__ out)
{
  const int count = *countp;
  const int m0 = blockIdx.y * 128;
  if (m0 >= count) return;
  const int h0 = blockIdx.x * 128;

  constexpr int BUF = 128 * 17;
  __shared__ int   toks[128];
  __shared__ float rws [128];
  __shared__ unsigned As[2 * BUF];
  __shared__ unsigned Bs[2 * BUF];

  const int tid = threadIdx.x;
  if (tid < 128) {
    const int m = m0 + tid;
    toks[tid] = (m < count) ? rowtok[m] : -1;
    rws[tid]  = (m < count) ? roww[m]   : 0.f;
  }
  __syncthreads();

  const int lane = tid & 31;
  const int wid  = tid >> 5;
  const int wM = (wid >> 1) * 32;
  const int wN = (wid &  1) * 64;

  const v8f zero8 = {0.f, 0.f, 0.f, 0.f, 0.f, 0.f, 0.f, 0.f};
  v8f acc[2][4];
  #pragma unroll
  for (int mi = 0; mi < 2; ++mi)
    #pragma unroll
    for (int ni = 0; ni < 4; ++ni) acc[mi][ni] = zero8;

  const int row  = tid >> 1;
  const int half = tid & 1;
  const unsigned* grow = (const unsigned*)(G + (size_t)(m0 + row) * INTERN) + half * 8;
  const float*    wrow = w2e + (size_t)(h0 + row) * INTERN + half * 16;
  unsigned* const dA = &As[row * 17 + half * 8];
  unsigned* const dB = &Bs[row * 17 + half * 8];

  TU8 rA; TF4 rB;
  fetch_u8(rA, grow);
  fetch_f16x(rB, wrow);
  stage_u8(rA, dA); stage_bf16(rB, dB);

  int buf = 0;
  for (int k0 = 32; k0 < INTERN; k0 += 32) {
    fetch_u8(rA, grow + (k0 >> 1));
    fetch_f16x(rB, wrow + k0);
    __builtin_prefetch((const void*)(wrow + k0 + 32), 0, 3);
    __syncthreads();
    mma_step_g2(As + buf * BUF, Bs + buf * BUF, wM, wN, lane, acc);
    const int nb = buf ^ 1;
    stage_u8(rA, dA + nb * BUF);
    stage_bf16(rB, dB + nb * BUF);
    buf = nb;
  }
  __syncthreads();
  mma_step_g2(As + buf * BUF, Bs + buf * BUF, wM, wN, lane, acc);

  // epilogue: scale by routing weight, scatter-add into out (2 contributions/token)
  const int lhalf = lane >> 4;
  const int lncol = lane & 15;
  #pragma unroll
  for (int mi = 0; mi < 2; ++mi) {
    #pragma unroll
    for (int ni = 0; ni < 4; ++ni) {
      FragC c; c.v = acc[mi][ni];
      #pragma unroll
      for (int v = 0; v < 8; ++v) {
        const int rowT = wM + mi * 16 + lhalf * 8 + v;
        const int tt = toks[rowT];
        if (tt >= 0) {
          const float val = rws[rowT] * c.f[v];
          const int col = h0 + wN + ni * 16 + lncol;
          __hip_atomic_fetch_add(&out[(size_t)tt * HID + col], val,
                                 __ATOMIC_RELAXED, __HIP_MEMORY_SCOPE_AGENT);
        }
      }
    }
  }
}

// ============ host launcher ============
extern "C" void kernel_launch(void* const* d_in, const int* in_sizes, int n_in,
                              void* d_out, int out_size, void* d_ws, size_t ws_size,
                              hipStream_t stream) {
  (void)in_sizes; (void)n_in; (void)out_size; (void)ws_size;
  const float* x  = (const float*)d_in[0];
  const float* gw = (const float*)d_in[1];
  const float* w1 = (const float*)d_in[2];
  const float* w3 = (const float*)d_in[3];
  const float* w2 = (const float*)d_in[4];
  float* out = (float*)d_out;

  char* p = (char*)d_ws;
  int*   topi   = (int*)p;    p += (size_t)T_TOK * 2 * sizeof(int);
  float* topw   = (float*)p;  p += (size_t)T_TOK * 2 * sizeof(float);
  int*   counts = (int*)p;    p += 256;
  int*   rowtok = (int*)p;    p += (size_t)NE * T_TOK * sizeof(int);
  float* roww   = (float*)p;  p += (size_t)NE * T_TOK * sizeof(float);
  unsigned short* G = (unsigned short*)p;  // T_TOK * INTERN bf16 (~112 MB), reused per expert

  moe_zero<<<4096, 256, 0, stream>>>(out, (long)T_TOK * HID, counts);
  moe_router<<<T_TOK / 8, 256, 0, stream>>>(x, gw, topi, topw);
  moe_build<<<T_TOK / 256, 256, 0, stream>>>(topi, topw, counts, rowtok, roww);

  for (int e = 0; e < NE; ++e) {
    moe_gemm1<<<dim3(INTERN / 128, T_TOK / 128), 256, 0, stream>>>(
        x, w1 + (size_t)e * INTERN * HID, w3 + (size_t)e * INTERN * HID,
        rowtok + (size_t)e * T_TOK, counts + e, G);
    moe_gemm2<<<dim3(HID / 128, T_TOK / 128), 256, 0, stream>>>(
        G, w2 + (size_t)e * HID * INTERN,
        rowtok + (size_t)e * T_TOK, roww + (size_t)e * T_TOK, counts + e, out);
  }
}
